// DenseGATLayer_627065225416
// MI455X (gfx1250) — compile-verified, matrix operand order checked by reading, not run on hardware
//
#include <hip/hip_runtime.h>
#include <hip/hip_bf16.h>

// CDNA5 wave32 WMMA types
typedef __attribute__((ext_vector_type(16))) __bf16 v16bf;
typedef __attribute__((ext_vector_type(8)))  float  v8f;

#define MTOT 8192   // B*N
#define DDIM 256    // D
#define NNODE 2048  // N
#define HH   4      // heads
#define DHH  64     // D/H

// ---------------------------------------------------------------- f32 -> bf16
__global__ void cvt_bf16_kernel(const float* __restrict__ in,
                                __bf16* __restrict__ out, int n) {
  int i = blockIdx.x * blockDim.x + threadIdx.x;
  if (i < n) out[i] = (__bf16)in[i];
}

// ------------------------------------------- GEMM1: h = x @ W^T  (bf16 WMMA)
// wave -> 16x64 output tile; K=256 fully unrolled -> 32 v_wmma per wave.
// Writes Hb row-major and HbT transposed (for contiguous attention B loads).
__global__ __launch_bounds__(256)
void gemm_xw_kernel(const __bf16* __restrict__ Xb, const __bf16* __restrict__ Wb,
                    __bf16* __restrict__ Hb, __bf16* __restrict__ HbT) {
  const int lane = threadIdx.x & 31;
  const int wave = blockIdx.x * (blockDim.x >> 5) + (threadIdx.x >> 5); // 0..2047
  const int mt   = wave >> 2;
  const int nblk = wave & 3;
  const int m0   = mt << 4;
  const int row  = m0 + (lane & 15);
  const int kg   = (lane >> 4) << 4; // lane-group K offset: 0 or 16

  v8f zero = {0.f,0.f,0.f,0.f,0.f,0.f,0.f,0.f};
  v8f acc[4] = {zero, zero, zero, zero};

#pragma unroll
  for (int k0 = 0; k0 < DDIM; k0 += 32) {
    v16bf a = *(const v16bf*)(Xb + row * DDIM + k0 + kg);
#pragma unroll
    for (int nn = 0; nn < 4; ++nn) {
      const int ncol = (nblk << 6) + (nn << 4) + (lane & 15);
      v16bf b = *(const v16bf*)(Wb + ncol * DDIM + k0 + kg);
      acc[nn] = __builtin_amdgcn_wmma_f32_16x16x32_bf16(
          false, a, false, b, (short)0, acc[nn], false, false);
    }
  }

  const int rbase = (lane < 16) ? 0 : 8;
#pragma unroll
  for (int nn = 0; nn < 4; ++nn) {
    const int ncol = (nblk << 6) + (nn << 4) + (lane & 15);
#pragma unroll
    for (int r = 0; r < 8; ++r) {
      const int m = m0 + r + rbase;
      __bf16 v = (__bf16)acc[nn][r];
      Hb[m * DDIM + ncol]  = v;
      HbT[ncol * MTOT + m] = v;
    }
  }
}

// ------------------------------- per-(row,head) attention projections src/dst
__global__ void srcdst_kernel(const __bf16* __restrict__ Hb,
                              const float* __restrict__ a_src,
                              const float* __restrict__ a_dst,
                              float* __restrict__ Src, float* __restrict__ Dst) {
  int idx = blockIdx.x * blockDim.x + threadIdx.x;
  if (idx >= MTOT * HH) return;
  int m = idx >> 2, h = idx & 3;
  const __bf16* hp = Hb + m * DDIM + h * DHH;
  const float* as = a_src + h * DHH;
  const float* ad = a_dst + h * DHH;
  float s = 0.f, d = 0.f;
#pragma unroll 8
  for (int t = 0; t < DHH; ++t) {
    float hv = (float)hp[t];
    s += hv * as[t];
    d += hv * ad[t];
  }
  Src[h * MTOT + m] = s;   // [H][M] layout -> contiguous loads in attention
  Dst[h * MTOT + m] = d;
}

// ------------------------- flash-style masked GAT attention, 1 wave / i-tile
// acc_tile(16x64) = softmax_j(mask(lrelu(src_i+dst_j))) @ h_j  via bf16 WMMA.
__global__ __launch_bounds__(128)
void gat_attn_kernel(const __bf16* __restrict__ HbT,
                     const float* __restrict__ Src, const float* __restrict__ Dst,
                     const int* __restrict__ Adj, __bf16* __restrict__ Ob) {
  const int lane = threadIdx.x & 31;
  const int wave = blockIdx.x * (blockDim.x >> 5) + (threadIdx.x >> 5); // 0..2047
  const int it    = wave & 127;
  const int head  = (wave >> 7) & 3;
  const int b     = wave >> 9;
  const int i0    = it << 4;
  const int mbase = b * NNODE;
  const int il    = lane & 15;
  const int jg    = lane >> 4;              // j half-chunk (wave32 pairing)
  const int rbase = (lane < 16) ? 0 : 8;

  const float my_src = Src[head * MTOT + mbase + i0 + il];
  const int*   arow  = Adj + (i0 + il) * NNODE;
  const float* dvec  = Dst + head * MTOT + mbase;

  v8f zero = {0.f,0.f,0.f,0.f,0.f,0.f,0.f,0.f};
  v8f acc[4] = {zero, zero, zero, zero};
  float m_run = -1e30f, l_run = 0.f;

  for (int j0 = 0; j0 < NNODE; j0 += 32) {
    const int jb = j0 + (jg << 4);
    float e[16];
    unsigned mb = 0u;
    float cmax = -1e30f;
#pragma unroll
    for (int t = 0; t < 16; ++t) {
      float ev = my_src + dvec[jb + t];
      ev = (ev >= 0.f) ? ev : 0.2f * ev;            // leaky_relu(0.2)
      if (arow[jb + t] != 0) { mb |= 1u << t; cmax = fmaxf(cmax, ev); }
      e[t] = ev;
    }
    // combine the two 16-lane halves of each row (wave32)
    cmax = fmaxf(cmax, __shfl_xor(cmax, 16));
    float m_new = fmaxf(m_run, cmax);
    float scale = __expf(m_run - m_new);
    float psum = 0.f;
    v16bf a;                                        // P tile lands in A layout
#pragma unroll
    for (int t = 0; t < 16; ++t) {
      float pv = (mb & (1u << t)) ? __expf(e[t] - m_new) : 0.f;
      psum += pv;
      a[t] = (__bf16)pv;
    }
    psum += __shfl_xor(psum, 16);
    l_run = l_run * scale + psum;
    m_run = m_new;
    // rescale accumulators: each lane holds rows rbase..rbase+7
#pragma unroll
    for (int r = 0; r < 8; ++r) {
      float sc = __shfl(scale, r + rbase);
#pragma unroll
      for (int nb = 0; nb < 4; ++nb) acc[nb][r] = acc[nb][r] * sc;
    }
    // acc += P(16x32) @ h_chunk(32x64): 4 WMMAs, contiguous loads from HbT
#pragma unroll
    for (int nb = 0; nb < 4; ++nb) {
      const int col = head * DHH + (nb << 4) + il;
      v16bf bm = *(const v16bf*)(HbT + col * MTOT + mbase + j0 + (jg << 4));
      acc[nb] = __builtin_amdgcn_wmma_f32_16x16x32_bf16(
          false, a, false, bm, (short)0, acc[nb], false, false);
    }
  }

  float linv = 1.f / l_run;
#pragma unroll
  for (int r = 0; r < 8; ++r) {
    float iv = __shfl(linv, r + rbase);
    const int m = mbase + i0 + r + rbase;
#pragma unroll
    for (int nb = 0; nb < 4; ++nb)
      Ob[m * DDIM + head * DHH + (nb << 4) + il] = (__bf16)(acc[nb][r] * iv);
  }
}

// ------------------------------------------- GEMM2: y = attn_out @ Wo^T (f32)
__global__ __launch_bounds__(256)
void gemm_ow_kernel(const __bf16* __restrict__ Ob, const __bf16* __restrict__ Wob,
                    float* __restrict__ Yf) {
  const int lane = threadIdx.x & 31;
  const int wave = blockIdx.x * (blockDim.x >> 5) + (threadIdx.x >> 5);
  const int mt   = wave >> 2;
  const int nblk = wave & 3;
  const int m0   = mt << 4;
  const int row  = m0 + (lane & 15);
  const int kg   = (lane >> 4) << 4;

  v8f zero = {0.f,0.f,0.f,0.f,0.f,0.f,0.f,0.f};
  v8f acc[4] = {zero, zero, zero, zero};

#pragma unroll
  for (int k0 = 0; k0 < DDIM; k0 += 32) {
    v16bf a = *(const v16bf*)(Ob + row * DDIM + k0 + kg);
#pragma unroll
    for (int nn = 0; nn < 4; ++nn) {
      const int ncol = (nblk << 6) + (nn << 4) + (lane & 15);
      v16bf b = *(const v16bf*)(Wob + ncol * DDIM + k0 + kg);
      acc[nn] = __builtin_amdgcn_wmma_f32_16x16x32_bf16(
          false, a, false, b, (short)0, acc[nn], false, false);
    }
  }

  const int rbase = (lane < 16) ? 0 : 8;
#pragma unroll
  for (int nn = 0; nn < 4; ++nn) {
    const int ncol = (nblk << 6) + (nn << 4) + (lane & 15);
#pragma unroll
    for (int r = 0; r < 8; ++r)
      Yf[(m0 + r + rbase) * DDIM + ncol] = acc[nn][r];
  }
}

// --------------------------------- residual + bias + LayerNorm (one row / WG)
__global__ __launch_bounds__(256)
void ln_kernel(const float* __restrict__ X, const float* __restrict__ Yf,
               const float* __restrict__ bo, const float* __restrict__ gamma,
               const float* __restrict__ beta, float* __restrict__ Out) {
  const int row = blockIdx.x, c = threadIdx.x;
  const float v = X[row * DDIM + c] + Yf[row * DDIM + c] + bo[c];
  __shared__ float s1[256];
  __shared__ float s2[256];
  s1[c] = v; s2[c] = v * v;
  __syncthreads();
  for (int s = 128; s > 0; s >>= 1) {
    if (c < s) { s1[c] += s1[c + s]; s2[c] += s2[c + s]; }
    __syncthreads();
  }
  const float mu  = s1[0] * (1.f / DDIM);
  const float var = s2[0] * (1.f / DDIM) - mu * mu;
  const float rs  = rsqrtf(var + 1e-5f);
  Out[row * DDIM + c] = (v - mu) * rs * gamma[c] + beta[c];
}

// ---------------------------------------------------------------------- host
extern "C" void kernel_launch(void* const* d_in, const int* in_sizes, int n_in,
                              void* d_out, int out_size, void* d_ws, size_t ws_size,
                              hipStream_t stream) {
  (void)in_sizes; (void)n_in; (void)out_size; (void)ws_size;
  const float* x     = (const float*)d_in[0];
  const int*   adj   = (const int*)d_in[1];
  const float* W     = (const float*)d_in[2];
  const float* a_src = (const float*)d_in[3];
  const float* a_dst = (const float*)d_in[4];
  const float* Wo    = (const float*)d_in[5];
  const float* bo    = (const float*)d_in[6];
  const float* gamma = (const float*)d_in[7];
  const float* beta  = (const float*)d_in[8];
  float* out = (float*)d_out;

  char* ws = (char*)d_ws;
  size_t off = 0;
  auto take = [&](size_t bytes) -> char* {
    char* p = ws + off;
    off += (bytes + 255) & ~(size_t)255;
    return p;
  };
  __bf16* Xb  = (__bf16*)take((size_t)MTOT * DDIM * 2);
  __bf16* Wb  = (__bf16*)take((size_t)DDIM * DDIM * 2);
  __bf16* Wob = (__bf16*)take((size_t)DDIM * DDIM * 2);
  __bf16* Hb  = (__bf16*)take((size_t)MTOT * DDIM * 2);
  __bf16* HbT = (__bf16*)take((size_t)MTOT * DDIM * 2);
  float*  Src = (float*)take((size_t)MTOT * HH * 4);
  float*  Dst = (float*)take((size_t)MTOT * HH * 4);
  __bf16* Ob  = (__bf16*)take((size_t)MTOT * DDIM * 2);
  float*  Yf  = (float*)take((size_t)MTOT * DDIM * 4);

  cvt_bf16_kernel<<<(MTOT * DDIM + 255) / 256, 256, 0, stream>>>(x,  Xb,  MTOT * DDIM);
  cvt_bf16_kernel<<<(DDIM * DDIM + 255) / 256, 256, 0, stream>>>(W,  Wb,  DDIM * DDIM);
  cvt_bf16_kernel<<<(DDIM * DDIM + 255) / 256, 256, 0, stream>>>(Wo, Wob, DDIM * DDIM);
  gemm_xw_kernel<<<256, 256, 0, stream>>>(Xb, Wb, Hb, HbT);
  srcdst_kernel<<<(MTOT * HH + 255) / 256, 256, 0, stream>>>(Hb, a_src, a_dst, Src, Dst);
  gat_attn_kernel<<<512, 128, 0, stream>>>(HbT, Src, Dst, adj, Ob);
  gemm_ow_kernel<<<256, 256, 0, stream>>>(Ob, Wob, Yf);
  ln_kernel<<<MTOT, 256, 0, stream>>>(x, Yf, bo, gamma, beta, out);
}
